// InvariantSlotAttention_26053271617984
// MI455X (gfx1250) — matrix operand order repaired and checked by reading, not compiled, not run on hardware
//
#include <hip/hip_runtime.h>
#include <hip/hip_bf16.h>

typedef unsigned short u16;
typedef unsigned int u32;
typedef __attribute__((ext_vector_type(16))) __bf16 v16bf;
typedef __attribute__((ext_vector_type(8)))  float  v8f;

#define BB 32
#define NPIX 16384      // H*W
#define NSLOT 8
#define NITER 3
#define SPLIT 16                        // workgroups per batch in attention
#define WSLICE (SPLIT * 8)              // wave-slices per batch = 128
#define PIXPERWAVE (NPIX / WSLICE)      // 128 pixels per wave
#define CHUNKS (PIXPERWAVE / 32)        // 4 chunks of 32 pixels

union FragA { uint4 q[2]; v16bf v; };

__device__ __forceinline__ u16 f2bf(float x) {
    u32 u = __builtin_bit_cast(u32, x);
    u32 r = (u + 0x7FFFu + ((u >> 16) & 1u)) >> 16;
    return (u16)r;
}
__device__ __forceinline__ float bf2f(u16 h) {
    u32 u = ((u32)h) << 16;
    return __builtin_bit_cast(float, u);
}

__device__ __forceinline__ v8f wmma_bf(v16bf a, v16bf b, v8f c) {
    return __builtin_amdgcn_wmma_f32_16x16x32_bf16(false, a, false, b, (short)0, c, false, false);
}

// A-matrix 16x32 bf16 fragment: row M = lane&15, dword j: K=(j<4?0:16)+8*hi+2*(j&3)
__device__ __forceinline__ v16bf load_a16(const u16* mat, int rowstride, int kbase, int lane) {
    int m = lane & 15, hi = lane >> 4;
    FragA f;
    const u16* p = mat + m * rowstride + kbase + hi * 8;
    f.q[0] = *(const uint4*)p;
    f.q[1] = *(const uint4*)(p + 16);
    return f.v;
}
// B-matrix 32x16 bf16 fragment: col N = n, dword j: K=(j<4?0:16)+8*hi+2*(j&3)
__device__ __forceinline__ v16bf load_b16(const u16* mat, int n, int rowstride, int kbase, int lane) {
    int hi = lane >> 4;
    FragA f;
    const u16* p = mat + (size_t)n * rowstride + kbase + hi * 8;
    f.q[0] = *(const uint4*)p;
    f.q[1] = *(const uint4*)(p + 16);
    return f.v;
}

// ---------------------------------------------------------------- init slots
__global__ void isa_init_slots(const float* __restrict__ noise, const float* __restrict__ mu,
                               const float* __restrict__ lsig, float* __restrict__ slots) {
    int i = blockIdx.x * 256 + threadIdx.x;   // 16384 total
    int d = i & 63;
    slots[i] = mu[d] + __expf(lsig[d]) * noise[i];
}

// ---------------------------------------------------------------- encoder: fused pos+LN+MLP+LN+K/V
__device__ __forceinline__ void loadWT(u16* dst, const float* w, int tid) {
    // dst[n*64+k] = bf16(w[k*64+n])  (transposed so B-frag K-pairs are contiguous)
    for (int i = tid; i < 4096; i += 256) {
        int n = i >> 6, k = i & 63;
        dst[i] = f2bf(w[k * 64 + n]);
    }
}

__global__ __launch_bounds__(256) void isa_encoder(
    const float* __restrict__ inputs, const float* __restrict__ absg,
    const float* __restrict__ pos_w, const float* __restrict__ pos_b,
    const float* __restrict__ encg, const float* __restrict__ encb,
    const float* __restrict__ w1,   const float* __restrict__ b1,
    const float* __restrict__ w2,   const float* __restrict__ b2,
    const float* __restrict__ inpg, const float* __restrict__ inpb,
    const float* __restrict__ wk,   const float* __restrict__ wv,
    u16* __restrict__ kbuf, u16* __restrict__ vt)
{
    __shared__ __align__(16) u16 wA[4096];
    __shared__ __align__(16) u16 wB[4096];
    __shared__ __align__(16) float prm[576];
    __shared__ __align__(16) u16 xa[8][1024];     // per-wave 16x64 bf16 tile
    __shared__ __align__(16) u16 vst[8192];       // 64 x 128 bf16 transpose stage

    int tid = threadIdx.x, lane = tid & 31, wave = tid >> 5;
    int l15 = lane & 15, hi = lane >> 4;
    int rowbase = blockIdx.x * 128 + wave * 16;
    int bidx = blockIdx.x >> 7;
    int nbase = (blockIdx.x & 127) * 128;

    loadWT(wA, w1, tid);
    loadWT(wB, w2, tid);
    if (tid < 64) {
        prm[tid]       = pos_w[tid];
        prm[64 + tid]  = pos_w[64 + tid];
        prm[128 + tid] = pos_b[tid];
        prm[192 + tid] = encg[tid];
        prm[256 + tid] = encb[tid];
        prm[320 + tid] = b1[tid];
        prm[384 + tid] = b2[tid];
        prm[448 + tid] = inpg[tid];
        prm[512 + tid] = inpb[tid];
    }
    __syncthreads();

    // ---- stage 1: input + pos-embed, LayerNorm(E), -> xa (bf16, [m][k])
    {
        int m2 = lane >> 1, hf = lane & 1;
        int grow = rowbase + m2;
        int pix = grow & (NPIX - 1);
        float gx = absg[pix * 2 + 0], gy = absg[pix * 2 + 1];
        float xv[32];
        const float4* ir = (const float4*)(inputs + (size_t)grow * 64 + hf * 32);
#pragma unroll
        for (int i = 0; i < 8; i++) {
            float4 t4 = ir[i];
            xv[4 * i] = t4.x; xv[4 * i + 1] = t4.y; xv[4 * i + 2] = t4.z; xv[4 * i + 3] = t4.w;
        }
        float s = 0.f, ss = 0.f;
#pragma unroll
        for (int e = 0; e < 32; e++) {
            int col = hf * 32 + e;
            float v = xv[e] + gx * prm[col] + gy * prm[64 + col] + prm[128 + col];
            xv[e] = v; s += v; ss += v * v;
        }
        s += __shfl_xor(s, 1); ss += __shfl_xor(ss, 1);
        float mean = s * (1.f / 64.f);
        float var = ss * (1.f / 64.f) - mean * mean;
        float rin = rsqrtf(var + 1e-5f);
        u16* xw = &xa[wave][m2 * 64 + hf * 32];
#pragma unroll
        for (int i = 0; i < 16; i++) {
            int col = hf * 32 + 2 * i;
            float a = (xv[2 * i] - mean) * rin * prm[192 + col] + prm[256 + col];
            float bb = (xv[2 * i + 1] - mean) * rin * prm[192 + col + 1] + prm[256 + col + 1];
            *(u32*)(xw + 2 * i) = (u32)f2bf(a) | ((u32)f2bf(bb) << 16);
        }
    }

    auto gemm4 = [&](const u16* wt, v8f* acc) {
        v16bf a0 = load_a16(&xa[wave][0], 64, 0, lane);
        v16bf a1 = load_a16(&xa[wave][0], 64, 32, lane);
#pragma unroll
        for (int t = 0; t < 4; t++) {
            v8f c = {0.f,0.f,0.f,0.f,0.f,0.f,0.f,0.f};
            c = wmma_bf(a0, load_b16(wt, t * 16 + l15, 64, 0, lane), c);
            c = wmma_bf(a1, load_b16(wt, t * 16 + l15, 64, 32, lane), c);
            acc[t] = c;
        }
    };
    auto store_acc = [&](v8f* acc, const float* bias, bool relu) {
#pragma unroll
        for (int t = 0; t < 4; t++) {
            float bb = bias ? bias[t * 16 + l15] : 0.f;
#pragma unroll
            for (int r = 0; r < 8; r++) {
                float f = acc[t][r] + bb;
                if (relu) f = fmaxf(f, 0.f);
                xa[wave][(r + 8 * hi) * 64 + t * 16 + l15] = f2bf(f);
            }
        }
    };

    // ---- stage 2: GEMM1 (+b1, relu) -> xa ; GEMM2 (+b2) -> xa
    {
        v8f acc[4];
        gemm4(wA, acc);
        store_acc(acc, &prm[320], true);
        gemm4(wB, acc);
        store_acc(acc, &prm[384], false);
    }
    // ---- stage 3: LayerNorm(D) over xa -> xa
    {
        int m2 = lane >> 1, hf = lane & 1;
        u16* xw = &xa[wave][m2 * 64 + hf * 32];
        float xv[32];
#pragma unroll
        for (int i = 0; i < 4; i++) {
            uint4 u = ((const uint4*)xw)[i];
            u32 ua[4] = {u.x, u.y, u.z, u.w};
#pragma unroll
            for (int j = 0; j < 4; j++) {
                xv[8 * i + 2 * j]     = bf2f((u16)(ua[j] & 0xffffu));
                xv[8 * i + 2 * j + 1] = bf2f((u16)(ua[j] >> 16));
            }
        }
        float s = 0.f, ss = 0.f;
#pragma unroll
        for (int e = 0; e < 32; e++) { s += xv[e]; ss += xv[e] * xv[e]; }
        s += __shfl_xor(s, 1); ss += __shfl_xor(ss, 1);
        float mean = s * (1.f / 64.f);
        float var = ss * (1.f / 64.f) - mean * mean;
        float rin = rsqrtf(var + 1e-5f);
#pragma unroll
        for (int i = 0; i < 16; i++) {
            int col = hf * 32 + 2 * i;
            float a = (xv[2 * i] - mean) * rin * prm[448 + col] + prm[512 + col];
            float bb = (xv[2 * i + 1] - mean) * rin * prm[448 + col + 1] + prm[512 + col + 1];
            *(u32*)(xw + 2 * i) = (u32)f2bf(a) | ((u32)f2bf(bb) << 16);
        }
    }

    // ---- stage 4: swap weights to wk/wv
    __syncthreads();
    loadWT(wA, wk, tid);
    loadWT(wB, wv, tid);
    __syncthreads();

    // ---- stage 5: K and V projections
    {
        v8f acck[4], accv[4];
        gemm4(wA, acck);
        gemm4(wB, accv);
        // k -> xa (bf16) -> coalesced global write [row][d]
        store_acc(acck, nullptr, false);
        {
            int m2 = lane >> 1, hf = lane & 1;
            const uint4* src = (const uint4*)&xa[wave][m2 * 64 + hf * 32];
            uint4* dst = (uint4*)(kbuf + (size_t)(rowbase + m2) * 64 + hf * 32);
#pragma unroll
            for (int i = 0; i < 4; i++) dst[i] = src[i];
        }
        // v -> vst transposed [d][wg_row]
#pragma unroll
        for (int t = 0; t < 4; t++) {
            int d = t * 16 + l15;
#pragma unroll
            for (int r2 = 0; r2 < 4; r2++) {
                u32 pk = (u32)f2bf(accv[t][2 * r2]) | ((u32)f2bf(accv[t][2 * r2 + 1]) << 16);
                *(u32*)&vst[d * 128 + wave * 16 + hi * 8 + 2 * r2] = pk;
            }
        }
    }
    __syncthreads();
    // ---- stage 6: cooperative coalesced write of v_t[b][d][n]
    {
        int d = tid >> 2, seg = tid & 3;
        const uint4* src = (const uint4*)&vst[d * 128 + seg * 32];
        uint4* dst = (uint4*)(vt + ((size_t)(bidx * 64 + d)) * NPIX + nbase + seg * 32);
#pragma unroll
        for (int i = 0; i < 4; i++) dst[i] = src[i];
    }
}

// ---------------------------------------------------------------- q = LN(slots) @ wq  (bf16 out)
__global__ __launch_bounds__(64) void isa_q(const float* __restrict__ slots,
                                            const float* __restrict__ sg, const float* __restrict__ sb,
                                            const float* __restrict__ wq, u16* __restrict__ qout) {
    __shared__ float sh[64], shn[64];
    int row = blockIdx.x, t = threadIdx.x;
    sh[t] = slots[row * 64 + t];
    __syncthreads();
    float s = 0.f, ss = 0.f;
    for (int j = 0; j < 64; j++) { float v = sh[j]; s += v; ss += v * v; }
    float mean = s * (1.f / 64.f);
    float var = ss * (1.f / 64.f) - mean * mean;
    float rin = rsqrtf(var + 1e-5f);
    shn[t] = (sh[t] - mean) * rin * sg[t] + sb[t];
    __syncthreads();
    float acc = 0.f;
    for (int j = 0; j < 64; j++) acc += shn[j] * wq[j * 64 + t];
    qout[row * 64 + t] = f2bf(acc);
}

// ---------------------------------------------------------------- attention: dots->softmax(slots)->updates
// grid = BB*SPLIT blocks x 256 threads; each wave streams PIXPERWAVE pixels.
__global__ __launch_bounds__(256) void isa_attn(const u16* __restrict__ kbuf, const u16* __restrict__ vt,
                                                const u16* __restrict__ qb, float* __restrict__ part) {
    __shared__ __align__(16) u16 attnlds[8][512];   // per-wave 16x32 bf16 (A-layout source)
    int tid = threadIdx.x, lane = tid & 31, wave = tid >> 5;
    int l15 = lane & 15, hi = lane >> 4;
    int b = blockIdx.x / SPLIT;
    int sp = blockIdx.x % SPLIT;
    int wslice = sp * 8 + wave;                     // 0..WSLICE-1
    const u16* kb = kbuf + (size_t)b * NPIX * 64;
    const u16* vb = vt + (size_t)b * 64 * NPIX;
    const u16* qrow = qb + (size_t)b * 512;

    FragA fq0, fq1;
    if (l15 < 8) {
        const u16* p = qrow + l15 * 64 + hi * 8;
        fq0.q[0] = *(const uint4*)p;        fq0.q[1] = *(const uint4*)(p + 16);
        fq1.q[0] = *(const uint4*)(p + 32); fq1.q[1] = *(const uint4*)(p + 48);
    } else {
        uint4 z; z.x = z.y = z.z = z.w = 0u;
        fq0.q[0] = fq0.q[1] = fq1.q[0] = fq1.q[1] = z;
    }

    v8f accu[4];
#pragma unroll
    for (int t = 0; t < 4; t++) accu[t] = (v8f){0.f,0.f,0.f,0.f,0.f,0.f,0.f,0.f};

    int pbase = wslice * PIXPERWAVE;
    for (int c = 0; c < CHUNKS; c++) {
        int p0 = pbase + c * 32;
        // prefetch next chunk's k rows (1 line/lane) and v rows (2 lines/lane)
        if (c + 1 < CHUNKS) {
            int pn = p0 + 32;
            __builtin_prefetch(kb + (size_t)(pn + lane) * 64, 0, 0);
            __builtin_prefetch(vb + (size_t)lane * NPIX + pn, 0, 0);
            __builtin_prefetch(vb + (size_t)(lane + 32) * NPIX + pn, 0, 0);
        }
#pragma unroll
        for (int hh = 0; hh < 2; hh++) {
            int n = p0 + hh * 16 + l15;
            v8f dv = {0.f,0.f,0.f,0.f,0.f,0.f,0.f,0.f};
            dv = wmma_bf(fq0.v, load_b16(kb, n, 64, 0, lane), dv);
            dv = wmma_bf(fq1.v, load_b16(kb, n, 64, 32, lane), dv);
            // per-pixel softmax over the 8 slots (all in this lane's 8 acc regs)
            float sc[8];
#pragma unroll
            for (int r = 0; r < 8; r++) sc[r] = dv[r] * 0.125f;
            float mx = sc[0];
#pragma unroll
            for (int r = 1; r < 8; r++) mx = fmaxf(mx, sc[r]);
            float sum = 0.f;
#pragma unroll
            for (int r = 0; r < 8; r++) { sc[r] = __expf(sc[r] - mx); sum += sc[r]; }
            float inv = 1.f / sum;
#pragma unroll
            for (int r = 0; r < 8; r++)
                attnlds[wave][(r + 8 * hi) * 32 + hh * 16 + l15] = f2bf(sc[r] * inv + 1e-8f);
        }
        // updates += attn(16x32) @ v(32x16 per d-tile); wave-local LDS transpose, in-order DS
        v16bf aa = load_a16(&attnlds[wave][0], 32, 0, lane);
#pragma unroll
        for (int t = 0; t < 4; t++)
            accu[t] = wmma_bf(aa, load_b16(vb + p0, t * 16 + l15, NPIX, 0, lane), accu[t]);
    }
    if (lane < 16) {
#pragma unroll
        for (int t = 0; t < 4; t++)
#pragma unroll
            for (int r = 0; r < 8; r++)
                part[(((size_t)b * WSLICE + wslice) * 8 + r) * 64 + t * 16 + l15] = accu[t][r];
    }
}

// ---------------------------------------------------------------- GRU + residual MLP on slots
__device__ __forceinline__ float sigm(float x) { return 1.f / (1.f + __expf(-x)); }

__global__ __launch_bounds__(64) void isa_gru(
    const float* __restrict__ slots_in, const float* __restrict__ part,
    const float* __restrict__ wih, const float* __restrict__ whh,
    const float* __restrict__ bih, const float* __restrict__ bhh,
    const float* __restrict__ ffg, const float* __restrict__ ffb,
    const float* __restrict__ mw1, const float* __restrict__ mb1,
    const float* __restrict__ mw2, const float* __restrict__ mb2,
    float* __restrict__ slots_out)
{
    __shared__ float uu[64], hh[64], s2[64], yv[64], hid[128];
    int row = blockIdx.x, t = threadIdx.x;
    int b = row >> 3, slot = row & 7;
    float u = 0.f;
#pragma unroll 8
    for (int w = 0; w < WSLICE; w++) u += part[(((size_t)b * WSLICE + w) * 8 + slot) * 64 + t];
    uu[t] = u;
    hh[t] = slots_in[row * 64 + t];
    __syncthreads();
    float gir = bih[t], giz = bih[64 + t], gin = bih[128 + t];
    float ghr = bhh[t], ghz = bhh[64 + t], ghn = bhh[128 + t];
    for (int j = 0; j < 64; j++) {
        float uj = uu[j], hj = hh[j];
        const float* wi = wih + j * 192;
        const float* wh = whh + j * 192;
        gir += uj * wi[t];       giz += uj * wi[64 + t];  gin += uj * wi[128 + t];
        ghr += hj * wh[t];       ghz += hj * wh[64 + t];  ghn += hj * wh[128 + t];
    }
    float r = sigm(gir + ghr);
    float z = sigm(giz + ghz);
    float ng = tanhf(gin + r * ghn);
    float h = hh[t];
    float sn = (1.f - z) * ng + z * h;
    s2[t] = sn;
    __syncthreads();
    float s = 0.f, ss = 0.f;
    for (int j = 0; j < 64; j++) { float v = s2[j]; s += v; ss += v * v; }
    float mean = s * (1.f / 64.f);
    float var = ss * (1.f / 64.f) - mean * mean;
    float rin = rsqrtf(var + 1e-5f);
    yv[t] = (sn - mean) * rin * ffg[t] + ffb[t];
    __syncthreads();
    float h0 = mb1[t], h1 = mb1[64 + t];
    for (int j = 0; j < 64; j++) {
        float yj = yv[j];
        const float* w = mw1 + j * 128;
        h0 += yj * w[t]; h1 += yj * w[64 + t];
    }
    hid[t] = fmaxf(h0, 0.f);
    hid[64 + t] = fmaxf(h1, 0.f);
    __syncthreads();
    float o = mb2[t];
    for (int j = 0; j < 128; j++) o += hid[j] * mw2[j * 64 + t];
    slots_out[row * 64 + t] = sn + o;
}

// ---------------------------------------------------------------- host launch
extern "C" void kernel_launch(void* const* d_in, const int* in_sizes, int n_in,
                              void* d_out, int out_size, void* d_ws, size_t ws_size,
                              hipStream_t stream) {
    const float* inputs  = (const float*)d_in[0];
    const float* absg    = (const float*)d_in[1];
    const float* noise   = (const float*)d_in[2];
    const float* mu      = (const float*)d_in[3];
    const float* lsig    = (const float*)d_in[4];
    const float* pos_w   = (const float*)d_in[5];
    const float* pos_b   = (const float*)d_in[6];
    const float* encg    = (const float*)d_in[7];
    const float* encb    = (const float*)d_in[8];
    const float* w1      = (const float*)d_in[9];
    const float* b1      = (const float*)d_in[10];
    const float* w2      = (const float*)d_in[11];
    const float* b2      = (const float*)d_in[12];
    const float* inpg    = (const float*)d_in[13];
    const float* inpb    = (const float*)d_in[14];
    const float* sg      = (const float*)d_in[15];
    const float* sb      = (const float*)d_in[16];
    const float* wq      = (const float*)d_in[17];
    const float* wk      = (const float*)d_in[18];
    const float* wv      = (const float*)d_in[19];
    const float* gwih    = (const float*)d_in[20];
    const float* gwhh    = (const float*)d_in[21];
    const float* gbih    = (const float*)d_in[22];
    const float* gbhh    = (const float*)d_in[23];
    const float* ffg     = (const float*)d_in[24];
    const float* ffb     = (const float*)d_in[25];
    const float* mw1     = (const float*)d_in[26];
    const float* mb1     = (const float*)d_in[27];
    const float* mw2     = (const float*)d_in[28];
    const float* mb2     = (const float*)d_in[29];

    char* ws = (char*)d_ws;
    const size_t KV = (size_t)BB * NPIX * 64 * 2;          // 67,108,864 bytes each
    u16*   kbuf  = (u16*)ws;
    u16*   vt    = (u16*)(ws + KV);
    float* slots = (float*)(ws + 2 * KV);
    u16*   qbuf  = (u16*)(ws + 2 * KV + 65536);
    float* part  = (float*)(ws + 2 * KV + 65536 + 32768);  // BB*WSLICE*8*64 f32 = 8 MB

    isa_init_slots<<<64, 256, 0, stream>>>(noise, mu, lsig, slots);
    isa_encoder<<<4096, 256, 0, stream>>>(inputs, absg, pos_w, pos_b, encg, encb,
                                          w1, b1, w2, b2, inpg, inpb, wk, wv, kbuf, vt);
    for (int it = 0; it < NITER; ++it) {
        isa_q<<<256, 64, 0, stream>>>(slots, sg, sb, wq, qbuf);
        isa_attn<<<BB * SPLIT, 256, 0, stream>>>(kbuf, vt, qbuf, part);
        float* outp = (it == NITER - 1) ? (float*)d_out : slots;
        isa_gru<<<256, 64, 0, stream>>>(slots, part, gwih, gwhh, gbih, gbhh,
                                        ffg, ffb, mw1, mb1, mw2, mb2, outp);
    }
}
